// SemIRLModel_20246475833838
// MI455X (gfx1250) — compile-verified, matrix operand order checked by reading, not run on hardware
//
#include <hip/hip_runtime.h>
#include <hip/hip_bf16.h>
#include <math.h>

#define GG   16      // grid_size
#define UU   4       // num_controls
#define CC   4       // num_classes
#define HID  64      // hidden channels
#define LBIG 1.0e6f  // "infinite" cost

typedef __attribute__((ext_vector_type(16))) __bf16 v16bf;
typedef __attribute__((ext_vector_type(8)))  __bf16 v8bf;
typedef __attribute__((ext_vector_type(4)))  __bf16 v4bf;
typedef __attribute__((ext_vector_type(8)))  float  v8f;

static __device__ __forceinline__ __bf16 f2bf(float x) { return (__bf16)x; }

// Fragment loader: a lane's 16 bf16 WMMA operand = two contiguous runs of 8
// K-values (run0 at +0, run1 at +16 elements), per the CDNA5 16-bit 16x32
// operand layout. base must point at run0 start (16B aligned).
static __device__ __forceinline__ v16bf frag_ld(const __bf16* base) {
    v8bf lo = *(const v8bf*)(base);
    v8bf hi = *(const v8bf*)(base + 16);
    return __builtin_shufflevector(lo, hi, 0, 1, 2, 3, 4, 5, 6, 7,
                                           8, 9, 10, 11, 12, 13, 14, 15);
}

__launch_bounds__(256)
__global__ void semirl_fused_kernel(const float* __restrict__ grid_cnt,  // (B,C,G,G)
                                    const int*   __restrict__ loc,       // (B,2)
                                    const int*   __restrict__ goal,      // (B,2)
                                    const float* __restrict__ theta,     // (C,G,G)
                                    const float* __restrict__ w1,        // (HID,C,3,3)
                                    const float* __restrict__ b1,        // (HID)
                                    const float* __restrict__ w2,        // (U,HID,3,3)
                                    const float* __restrict__ b2,        // (U)
                                    float*       __restrict__ out,       // logits then probs
                                    int batch)
{
    // ---- LDS (~98 KB/block) ----
    __shared__ __align__(16) float  s_gc[CC * GG * GG];            // staged grid_cnt slice
    __shared__ __align__(16) __bf16 s_prob[(GG + 2) * (GG + 2) * CC]; // [r][c][ch], zero halo
    __shared__ __align__(16) __bf16 s_A1[256][64];                 // conv1 im2col, K=tap*4+ch pad->64
    __shared__ __align__(16) __bf16 s_w1b[64][64];                 // conv1 W, [n][k]
    __shared__ __align__(16) __bf16 s_h[(GG + 2) * (GG + 2)][HID]; // [pixel][hid], zero halo
    __shared__ __align__(16) __bf16 s_w2b[5][576];                 // conv2 W, [n][k=tap*64+hid], row4=0
    __shared__ float  s_cost[UU][GG][GG];
    __shared__ float  s_g[GG + 2][GG + 2];
    __shared__ float  s_q[UU];

    const int tid  = threadIdx.x;       // 0..255, 8 waves
    const int b    = blockIdx.x;
    const int wave = tid >> 5;
    const int lane = tid & 31;
    const int half = lane >> 4;         // selects K run offset (8*half)
    const int mn   = lane & 15;         // A-row / B-col within 16x16 tile

    // Warm L2/WGP$ for the shared small tensors (global_prefetch_b8).
    __builtin_prefetch(theta, 0, 1);
    __builtin_prefetch(w1, 0, 1);
    __builtin_prefetch(w2, 0, 1);

    // ---- stage 0a: async-copy this block's grid_cnt slice (4 KB) into LDS ----
    {
        const float* src = grid_cnt + (size_t)b * (CC * GG * GG) + tid * 4;
        unsigned lds_addr = (unsigned)(uintptr_t)&s_gc[tid * 4];
        unsigned long long ga = (unsigned long long)(uintptr_t)src;
        asm volatile("global_load_async_to_lds_b128 %0, %1, off"
                     :: "v"(lds_addr), "v"(ga) : "memory");
    }

    // ---- stage 0b: zero prob halo, stage weights into LDS in B-matrix layout ----
    {
        v4bf z4 = {};
        for (int i = tid; i < (int)(sizeof(s_prob) / 8); i += 256)
            ((v4bf*)s_prob)[i] = z4;
    }
    for (int i = tid; i < 64 * 64; i += 256) {          // w1: [n][k], k = tap*4+ch, pad k>=36
        int n = i >> 6, k = i & 63;
        float v = 0.f;
        if (k < 36) {
            int tap = k >> 2, ch = k & 3;
            v = w1[((n * CC + ch) * 3 + tap / 3) * 3 + (tap % 3)];
        }
        ((__bf16*)s_w1b)[i] = f2bf(v);
    }
    for (int n = 0; n < 5; ++n)                          // w2: [n][k], k = tap*64+hid, row4 = 0
        for (int k = tid; k < 576; k += 256) {
            int tap = k >> 6, hid = k & 63;
            float v = (n < UU) ? w2[((n * HID + hid) * 3 + tap / 3) * 3 + (tap % 3)] : 0.f;
            s_w2b[n][k] = f2bf(v);
        }

    asm volatile("s_wait_asynccnt 0" ::: "memory");
    __syncthreads();

    // ---- stage 1: per-pixel class softmax (one thread per pixel) ----
    {
        const int r = tid >> 4, c = tid & 15;
        float x[CC], m = -1e30f;
        #pragma unroll
        for (int ch = 0; ch < CC; ++ch) {
            x[ch] = s_gc[ch * 256 + tid] * theta[(ch * GG + r) * GG + c];
            m = fmaxf(m, x[ch]);
        }
        float s = 0.f, e[CC];
        #pragma unroll
        for (int ch = 0; ch < CC; ++ch) { e[ch] = __expf(x[ch] - m); s += e[ch]; }
        const float inv = 1.f / s;
        #pragma unroll
        for (int ch = 0; ch < CC; ++ch)
            s_prob[((1 + r) * (GG + 2) + (1 + c)) * CC + ch] = f2bf(e[ch] * inv);
    }
    __syncthreads();

    // ---- stage 2a: zero h (halo) + build conv1 im2col panel (one pixel/thread) ----
    {
        float4 z = make_float4(0.f, 0.f, 0.f, 0.f);
        for (int i = tid; i < (int)(sizeof(s_h) / 16); i += 256)
            ((float4*)s_h)[i] = z;

        const int pr = tid >> 4, pc = tid & 15;
        #pragma unroll
        for (int tap = 0; tap < 9; ++tap) {
            const int dr = tap / 3, dc = tap % 3;        // compile-time
            v4bf t = *(const v4bf*)&s_prob[((pr + dr) * (GG + 2) + (pc + dc)) * CC];
            *(v4bf*)&s_A1[tid][tap * 4] = t;
        }
        v4bf z4 = {};
        #pragma unroll
        for (int k = 36; k < 64; k += 4) *(v4bf*)&s_A1[tid][k] = z4;
    }
    __syncthreads();

    // ---- stage 2b: conv1 (4->64) WMMA GEMM: M=256, K=64(pad from 36), N=64 ----
    float bias1[4];
    #pragma unroll
    for (int nt = 0; nt < 4; ++nt) bias1[nt] = b1[nt * 16 + mn];

    for (int mt = 2 * wave; mt <= 2 * wave + 1; ++mt) {
        const int p = mt * 16 + mn;                      // A-row pixel for this lane
        v16bf a0 = frag_ld(&s_A1[p][0  + 8 * half]);
        v16bf a1 = frag_ld(&s_A1[p][32 + 8 * half]);
        #pragma unroll
        for (int nt = 0; nt < 4; ++nt) {
            const int col = nt * 16 + mn;                // B-col channel for this lane
            v8f acc = {};
            acc = __builtin_amdgcn_wmma_f32_16x16x32_bf16(
                      false, a0, false, frag_ld(&s_w1b[col][0 + 8 * half]),
                      (short)0, acc, false, false);
            acc = __builtin_amdgcn_wmma_f32_16x16x32_bf16(
                      false, a1, false, frag_ld(&s_w1b[col][32 + 8 * half]),
                      (short)0, acc, false, false);
            // D: VGPR j -> M = j + 8*half, N = lane&15
            #pragma unroll
            for (int j = 0; j < 8; ++j) {
                const int pp = mt * 16 + j + 8 * half;
                const float v = fmaxf(acc[j] + bias1[nt], 0.f);  // bias + ReLU
                s_h[(1 + (pp >> 4)) * (GG + 2) + 1 + (pp & 15)][col] = f2bf(v);
            }
        }
    }
    __syncthreads();

    // ---- stage 3: conv2 (64->4) WMMA GEMM: M=256, K=576 (tap-major), N=4(pad 16) ----
    {
        const int colw  = (mn < UU) ? mn : 4;            // zero row for padded cols
        const float b2v = (mn < UU) ? b2[mn] : 0.f;
        for (int mt = 2 * wave; mt <= 2 * wave + 1; ++mt) {
            const int p = mt * 16 + mn;
            const int pr = p >> 4, pc = p & 15;
            v8f acc = {};
            #pragma unroll
            for (int ks = 0; ks < 18; ++ks) {
                const int tap = ks >> 1;                 // compile-time per unrolled step
                const int dr = tap / 3, dc = tap % 3;
                const __bf16* abase =
                    &s_h[(pr + dr) * (GG + 2) + (pc + dc)][(ks & 1) * 32 + 8 * half];
                acc = __builtin_amdgcn_wmma_f32_16x16x32_bf16(
                          false, frag_ld(abase),
                          false, frag_ld(&s_w2b[colw][ks * 32 + 8 * half]),
                          (short)0, acc, false, false);
            }
            #pragma unroll
            for (int j = 0; j < 8; ++j) {
                const int pp = mt * 16 + j + 8 * half;
                if (mn < UU) {
                    const float x  = acc[j] + b2v;
                    const float sp = fmaxf(x, 0.f) + log1pf(__expf(-fabsf(x)));  // softplus
                    s_cost[mn][pp >> 4][pp & 15] = sp + 1e-6f;
                }
            }
        }
    }

    // ---- stage 4: Bellman value iteration, 256 Jacobi sweeps in LDS ----
    for (int i = tid; i < (GG + 2) * (GG + 2); i += 256)
        ((float*)s_g)[i] = LBIG;                         // halo = L (off-grid)
    __syncthreads();
    if (tid == 0) {
        const int gr = goal[2 * b], gc = goal[2 * b + 1];
        s_g[1 + gr][1 + gc] = 0.f;
    }
    __syncthreads();

    const int r = tid >> 4, c = tid & 15;
    const float c0 = s_cost[0][r][c], c1 = s_cost[1][r][c];
    const float c2 = s_cost[2][r][c], c3 = s_cost[3][r][c];
    for (int it = 0; it < GG * GG; ++it) {
        float best = s_g[1 + r][1 + c];
        best = fminf(best, c0 + s_g[r    ][1 + c]);      // u0: (-1, 0)
        best = fminf(best, c1 + s_g[2 + r][1 + c]);      // u1: (+1, 0)
        best = fminf(best, c2 + s_g[1 + r][c    ]);      // u2: ( 0,-1)
        best = fminf(best, c3 + s_g[1 + r][2 + c]);      // u3: ( 0,+1)
        __syncthreads();
        s_g[1 + r][1 + c] = best;
        __syncthreads();
    }

    // ---- stage 5: Q(start,u), logits, softmax over controls ----
    if (tid < UU) {
        const int lr = loc[2 * b], lc = loc[2 * b + 1];
        const int dr = (tid == 0) ? -1 : (tid == 1) ? 1 : 0;
        const int dc = (tid == 2) ? -1 : (tid == 3) ? 1 : 0;
        const float q = s_cost[tid][lr][lc] + s_g[1 + lr + dr][1 + lc + dc];
        s_q[tid] = -q;
    }
    __syncthreads();
    if (tid < UU) {
        const float m = fmaxf(fmaxf(s_q[0], s_q[1]), fmaxf(s_q[2], s_q[3]));
        const float s = __expf(s_q[0] - m) + __expf(s_q[1] - m) +
                        __expf(s_q[2] - m) + __expf(s_q[3] - m);
        const float logit = s_q[tid];
        out[b * UU + tid]              = logit;                    // logits
        out[batch * UU + b * UU + tid] = __expf(logit - m) / s;    // probs
    }
}

extern "C" void kernel_launch(void* const* d_in, const int* in_sizes, int n_in,
                              void* d_out, int out_size, void* d_ws, size_t ws_size,
                              hipStream_t stream) {
    const float* grid_cnt = (const float*)d_in[0];
    const int*   loc      = (const int*)  d_in[1];
    const int*   goal     = (const int*)  d_in[2];
    const float* theta    = (const float*)d_in[3];
    const float* w1       = (const float*)d_in[4];
    const float* b1       = (const float*)d_in[5];
    const float* w2       = (const float*)d_in[6];
    const float* b2       = (const float*)d_in[7];
    float*       out      = (float*)d_out;

    const int batch = in_sizes[1] / 2;   // loc_batch is (B,2)
    semirl_fused_kernel<<<batch, 256, 0, stream>>>(
        grid_cnt, loc, goal, theta, w1, b1, w2, b2, out, batch);
}
